// FeaturePropagation_24781961298016
// MI455X (gfx1250) — compile-verified, matrix operand order checked by reading, not run on hardware
//
#include <hip/hip_runtime.h>
#include <hip/hip_bf16.h>
#include <stdint.h>

typedef __attribute__((ext_vector_type(16))) _Float16     v16h;
typedef __attribute__((ext_vector_type(8)))  float        v8f;
typedef __attribute__((ext_vector_type(4)))  unsigned int u32x4;
typedef __attribute__((ext_vector_type(8)))  int          i32x8;
typedef __attribute__((ext_vector_type(4)))  int          i32x4;

#define BB    4
#define NN    16384
#define NP    4096
#define C1C   64
#define C2C   128
#define INCH  192
#define H1    256
#define H2    128
#define MROWS (BB*NN)    // 65536
#define BN_EPS 1e-5f

#if defined(__has_builtin)
#if __has_builtin(__builtin_amdgcn_tensor_load_to_lds) && \
    __has_builtin(__builtin_amdgcn_s_wait_tensorcnt)
#define HAS_TDM 1
#endif
#endif

union AFrag { uint32_t u[8]; v16h h; };
union HPack { _Float16 h[2]; uint32_t u; };

// ---------------------------------------------------------------------------
// Kernel 1: 3-NN inverse-distance interpolation + concat [points1 | interp]
// into an f16 activation matrix xh [MROWS x 192]. xyz2 staged via LDS tiles.
// ---------------------------------------------------------------------------
__global__ __launch_bounds__(256)
void k_interp_concat(const float* __restrict__ xyz1, const float* __restrict__ xyz2,
                     const float* __restrict__ points1, const float* __restrict__ points2,
                     _Float16* __restrict__ xh)
{
  __shared__ float sx[1024*3];
  const int tid = threadIdx.x;
  const int row = blockIdx.x*256 + tid;     // blocks never straddle a batch
  const int b   = row / NN;
  const float px = xyz1[row*3+0];
  const float py = xyz1[row*3+1];
  const float pz = xyz1[row*3+2];

  float d0=3.4e38f, d1=3.4e38f, d2v=3.4e38f;
  int   i0=0, i1=0, i2=0;

  for (int t = 0; t < NP; t += 1024) {
    float tmp[12];
    #pragma unroll
    for (int i = 0; i < 12; ++i) tmp[i] = xyz2[(b*NP + t)*3 + tid + i*256];
    __syncthreads();
    #pragma unroll
    for (int i = 0; i < 12; ++i) sx[tid + i*256] = tmp[i];
    __syncthreads();
    #pragma unroll 4
    for (int j = 0; j < 1024; ++j) {
      float dx = px - sx[j*3+0];
      float dy = py - sx[j*3+1];
      float dz = pz - sx[j*3+2];
      float d  = dx*dx + dy*dy + dz*dz;
      int   m  = t + j;
      if (d < d2v) {
        if (d < d1) {
          d2v = d1; i2 = i1;
          if (d < d0) { d1 = d0; i1 = i0; d0 = d; i0 = m; }
          else        { d1 = d;  i1 = m; }
        } else { d2v = d; i2 = m; }
      }
    }
    __syncthreads();
  }

  float r0 = 1.0f/(d0 +1e-8f), r1 = 1.0f/(d1 +1e-8f), r2 = 1.0f/(d2v+1e-8f);
  float inv = 1.0f/(r0+r1+r2);
  float w0 = r0*inv, w1 = r1*inv, w2 = r2*inv;

  const float* p0 = points2 + ((size_t)b*NP + i0)*C2C;
  const float* p1 = points2 + ((size_t)b*NP + i1)*C2C;
  const float* p2 = points2 + ((size_t)b*NP + i2)*C2C;
  const float* q  = points1 + (size_t)row*C1C;
  _Float16* xrow  = xh + (size_t)row*INCH;

  #pragma unroll 8
  for (int c = 0; c < C1C; ++c) xrow[c] = (_Float16)q[c];
  #pragma unroll 8
  for (int c = 0; c < C2C; ++c) {
    float v = w0*p0[c] + w1*p1[c] + w2*p2[c];
    xrow[C1C+c] = (_Float16)v;
  }
}

// ---------------------------------------------------------------------------
// WMMA GEMM: Y[M x N] = A[M x K](f16) * W[N x K]^T (f32->f16) + bias.
// 256 threads = 8 waves; tile 128(M) x 32(N); K stepped by 32.
// A tile is DMA'd into LDS by the Tensor Data Mover (TDM) with hardware
// padding (16 dwords + 1 pad dword per row => 34-halfword stride, conflict-
// free since 17 is coprime with 64 banks); wave 0 issues the descriptor and
// waits on TENSORcnt. B tile is staged pair-packed so every fragment element
// is one b32 LDS load.
//   A/B 16-bit fragment layout: lane = half*16 + l;
//   dword p -> K-pair index kd = 4*half + (p&3) + 8*(p>>2)
//   C/D layout: lane holds N=l, M = 8*half + j (j = vgpr 0..7).
// ---------------------------------------------------------------------------
template<bool OUT_F16>
__global__ __launch_bounds__(256)
void k_gemm_wmma(const _Float16* __restrict__ A, const float* __restrict__ W,
                 const float* __restrict__ bias, void* __restrict__ Yout,
                 int M, int N, int K)
{
  __shared__ _Float16 As[128*34];   // 128 rows x (32 K + 2 pad)
  __shared__ uint32_t Bp[16*32];    // 16 K-pairs x 32 N, pair-packed f16

  const int tid  = threadIdx.x;
  const int lane = tid & 31;
  const int wid  = tid >> 5;        // 0..7
  const int half = lane >> 4;       // 0/1
  const int l    = lane & 15;
  const int mBase = blockIdx.x * 128;
  const int nBase = blockIdx.y * 32;
  const int rBase = wid * 16;
  const uint32_t* As32 = (const uint32_t*)As;

  v8f acc0 = {}; v8f acc1 = {};

  for (int k0 = 0; k0 < K; k0 += 32) {
#if HAS_TDM
    if (wid == 0) {
      // --- Tensor DMA descriptor: 2D tile 32(x) x 128(y), 2-byte elements,
      //     row stride K elements, LDS pad 1 dword per 16 dwords. ---
      const uint64_t ga  = (uint64_t)(uintptr_t)(A + (size_t)mBase*K + k0);
      const uint32_t lds = (uint32_t)(uintptr_t)&As[0];  // low 32b = LDS offset
      const uint32_t Ku  = (uint32_t)K, Mu = (uint32_t)M;
      u32x4 g0;
      g0[0] = 1u;                                            // count=1
      g0[1] = lds;                                           // lds_addr
      g0[2] = (uint32_t)ga;                                  // global_addr lo
      g0[3] = (uint32_t)((ga >> 32) & 0x1FFFFFFu) | (2u<<30);// addr hi | type=2
      i32x8 g1;
      g1[0] = (int)(0x00010000u | (1u<<20) | (3u<<22));      // 2B elems, pad 1dw/16dw
      g1[1] = (int)((Ku & 0xFFFFu) << 16);                   // tensor_dim0 lo16
      g1[2] = (int)((Ku >> 16) | ((Mu & 0xFFFFu) << 16));    // dim0 hi | dim1 lo
      g1[3] = (int)((Mu >> 16) | (32u << 16));               // dim1 hi | tile_dim0=32
      g1[4] = (int)128u;                                     // tile_dim1=128
      g1[5] = (int)Ku;                                       // dim0_stride lo32
      g1[6] = 0; g1[7] = 0;
      i32x4 g2 = {0,0,0,0};                                  // dims 2/3: unused
      i32x4 g3 = {0,0,0,0};                                  // dim 4: unused
      i32x8 g4 = {0,0,0,0,0,0,0,0};                          // extra group (unused)
      __builtin_amdgcn_tensor_load_to_lds(g0, g1, g2, g3, g4, 0);
      __builtin_amdgcn_s_wait_tensorcnt(0);
    }
#else
    {
      // Fallback: two-phase register-staged copy (all loads, then all stores).
      uint32_t va[8];
      #pragma unroll
      for (int i = 0; i < 8; ++i) {
        int pi = tid + i*256;
        int r  = pi >> 4;
        int c2 = (pi & 15) << 1;
        va[i] = *(const uint32_t*)(A + (size_t)(mBase + r)*K + k0 + c2);
      }
      #pragma unroll
      for (int i = 0; i < 8; ++i) {
        int pi = tid + i*256;
        ((uint32_t*)As)[(pi >> 4)*17 + (pi & 15)] = va[i];
      }
    }
#endif
    // Stage B tile pair-packed: Bp[kd*32+n] = {f16 W[n][2kd], f16 W[n][2kd+1]}
    {
      HPack t2[2];
      #pragma unroll
      for (int i = 0; i < 2; ++i) {
        int idx = tid + i*256;            // 0..511
        int nn  = idx & 31;
        int kp  = idx >> 5;               // 0..15
        const float* wp = W + (size_t)(nBase + nn)*K + k0 + 2*kp;
        t2[i].h[0] = (_Float16)wp[0];
        t2[i].h[1] = (_Float16)wp[1];
      }
      #pragma unroll
      for (int i = 0; i < 2; ++i) {
        int idx = tid + i*256;
        Bp[(idx >> 5)*32 + (idx & 31)] = t2[i].u;
      }
    }
    __syncthreads();

    AFrag a, b0, b1;
    #pragma unroll
    for (int p = 0; p < 8; ++p) {
      int kd = 4*half + (p & 3) + 8*(p >> 2);     // K-pair index 0..15
      a.u[p]  = As32[(rBase + l)*17 + kd];
      b0.u[p] = Bp[kd*32 + l];
      b1.u[p] = Bp[kd*32 + 16 + l];
    }
    acc0 = __builtin_amdgcn_wmma_f32_16x16x32_f16(false, a.h, false, b0.h,
                                                  (short)0, acc0, false, false);
    acc1 = __builtin_amdgcn_wmma_f32_16x16x32_f16(false, a.h, false, b1.h,
                                                  (short)0, acc1, false, false);
    __syncthreads();
  }

  #pragma unroll
  for (int j = 0; j < 8; ++j) {
    int row  = mBase + rBase + 8*half + j;
    int col0 = nBase + l;
    int col1 = nBase + 16 + l;
    float v0 = acc0[j] + bias[col0];
    float v1 = acc1[j] + bias[col1];
    if (OUT_F16) {
      _Float16* Y = (_Float16*)Yout;
      Y[(size_t)row*N + col0] = (_Float16)v0;
      Y[(size_t)row*N + col1] = (_Float16)v1;
    } else {
      float* Y = (float*)Yout;
      Y[(size_t)row*N + col0] = v0;
      Y[(size_t)row*N + col1] = v1;
    }
  }
}

// ---------------------------------------------------------------------------
// Per-channel BN stats (biased var, training mode) folded into scale/shift.
// One block per channel; fixed-order tree reduction => deterministic.
// ---------------------------------------------------------------------------
template<typename T>
__global__ __launch_bounds__(256)
void k_bn_stats_fold(const T* __restrict__ Y, int Mrows, int C,
                     const float* __restrict__ g, const float* __restrict__ beta,
                     float* __restrict__ scale, float* __restrict__ shift)
{
  __shared__ float ssum[256];
  __shared__ float ssq[256];
  const int c   = blockIdx.x;
  const int tid = threadIdx.x;
  float s = 0.f, s2 = 0.f;
  for (int r = tid; r < Mrows; r += 256) {
    float v = (float)Y[(size_t)r*C + c];
    s += v; s2 += v*v;
  }
  ssum[tid] = s; ssq[tid] = s2;
  __syncthreads();
  for (int o = 128; o > 0; o >>= 1) {
    if (tid < o) { ssum[tid] += ssum[tid+o]; ssq[tid] += ssq[tid+o]; }
    __syncthreads();
  }
  if (tid == 0) {
    float mean = ssum[0] / (float)Mrows;
    float var  = ssq[0]  / (float)Mrows - mean*mean;
    float sc   = g[c] * rsqrtf(var + BN_EPS);
    scale[c] = sc;
    shift[c] = beta[c] - mean*sc;
  }
}

// ---------------------------------------------------------------------------
// Apply y = relu(y*scale[c] + shift[c]) in place. C is a power of two.
// ---------------------------------------------------------------------------
template<typename T>
__global__ __launch_bounds__(256)
void k_bn_apply_relu(T* __restrict__ Y, size_t total, int C,
                     const float* __restrict__ scale, const float* __restrict__ shift)
{
  size_t i      = (size_t)blockIdx.x*256 + threadIdx.x;
  size_t stride = (size_t)gridDim.x*256;
  for (; i < total; i += stride) {
    int   c = (int)i & (C - 1);
    float v = (float)Y[i];
    v = v*scale[c] + shift[c];
    Y[i] = (T)(v > 0.f ? v : 0.f);
  }
}

// ---------------------------------------------------------------------------
// Launch. Workspace (~59 MB): xh f16 [65536x192], y1h f16 [65536x256],
// BN scale/shift. Final layer writes raw f32 into d_out, BN+ReLU in place.
// ---------------------------------------------------------------------------
extern "C" void kernel_launch(void* const* d_in, const int* in_sizes, int n_in,
                              void* d_out, int out_size, void* d_ws, size_t ws_size,
                              hipStream_t stream)
{
  (void)in_sizes; (void)n_in; (void)out_size; (void)ws_size;

  const float* xyz1    = (const float*)d_in[0];
  const float* xyz2    = (const float*)d_in[1];
  const float* points1 = (const float*)d_in[2];
  const float* points2 = (const float*)d_in[3];
  const float* W1  = (const float*)d_in[4];
  const float* b1  = (const float*)d_in[5];
  const float* g1  = (const float*)d_in[6];
  const float* be1 = (const float*)d_in[7];
  const float* W2  = (const float*)d_in[8];
  const float* b2  = (const float*)d_in[9];
  const float* g2  = (const float*)d_in[10];
  const float* be2 = (const float*)d_in[11];

  _Float16* xh  = (_Float16*)d_ws;
  _Float16* y1h = xh + (size_t)MROWS*INCH;
  float*    ss  = (float*)(y1h + (size_t)MROWS*H1);
  float* scale1 = ss;
  float* shift1 = ss + H1;
  float* scale2 = ss + 2*H1;
  float* shift2 = ss + 2*H1 + H2;
  float* yout   = (float*)d_out;

  k_interp_concat<<<MROWS/256, 256, 0, stream>>>(xyz1, xyz2, points1, points2, xh);

  k_gemm_wmma<true><<<dim3(MROWS/128, H1/32), 256, 0, stream>>>(
      xh, W1, b1, (void*)y1h, MROWS, H1, INCH);
  k_bn_stats_fold<_Float16><<<H1, 256, 0, stream>>>(y1h, MROWS, H1, g1, be1, scale1, shift1);
  k_bn_apply_relu<_Float16><<<4096, 256, 0, stream>>>(y1h, (size_t)MROWS*H1, H1, scale1, shift1);

  k_gemm_wmma<false><<<dim3(MROWS/128, H2/32), 256, 0, stream>>>(
      y1h, W2, b2, (void*)yout, MROWS, H2, H1);
  k_bn_stats_fold<float><<<H2, 256, 0, stream>>>(yout, MROWS, H2, g2, be2, scale2, shift2);
  k_bn_apply_relu<float><<<4096, 256, 0, stream>>>(yout, (size_t)MROWS*H2, H2, scale2, shift2);
}
